// XLRelPartialLearnableMultiHeadAttn_23072564314384
// MI455X (gfx1250) — matrix operand order repaired and checked by reading, not compiled, not run on hardware
//
#include <hip/hip_runtime.h>
#include <stdint.h>

// ---------------- problem constants ----------------
#define NH    16
#define DM    1024
#define DH    64
#define QL    1024
#define ML    1024
#define KL    2048     // QL + ML
#define BB    4
#define SCALE 0.125f   // 1/sqrt(64)
#define NEG_BIG (-1.0e30f)

// ---------------- CDNA5 WMMA types ----------------
typedef __bf16 bf16x16 __attribute__((ext_vector_type(16)));
typedef float  f32x8   __attribute__((ext_vector_type(8)));

union FragB { bf16x16 v; uint4 q[2]; };
union AccF  { f32x8   v; float f[8]; };

__device__ __forceinline__ uint16_t f32_to_bf16_bits(float x) {
  union { float f; uint32_t u; } c; c.f = x;
  uint32_t u = c.u;
  uint32_t r = u + 0x7FFFu + ((u >> 16) & 1u);   // round-to-nearest-even
  return (uint16_t)(r >> 16);
}

__device__ __forceinline__ f32x8 wmma_bf16(FragB a, FragB b, f32x8 c) {
  // v_wmma_f32_16x16x32_bf16  D = A(16x32) x B(32x16) + C
  return __builtin_amdgcn_wmma_f32_16x16x32_bf16(
      false, a.v, false, b.v, (short)0, c, false, false);
}

// ---------------- CDNA5 async copy (global -> LDS), tracked by ASYNCcnt ----------
// one instruction moves 16 bytes per lane; LDS byte address in VGPR (low 32 bits
// of the generic pointer ARE the LDS address per ISA aperture rules)
#define ASYNC_LD16(lds32, gaddr64)                                          \
  asm volatile("global_load_async_to_lds_b128 %0, %1, off"                  \
               :: "v"(lds32), "v"(gaddr64) : "memory")

__device__ __forceinline__ void wait_async_le(int) {}
__device__ __forceinline__ void wait_async_2() { asm volatile("s_wait_asynccnt 0x2" ::: "memory"); }
__device__ __forceinline__ void wait_async_8() { asm volatile("s_wait_asynccnt 0x8" ::: "memory"); }
__device__ __forceinline__ void wait_async_0() { asm volatile("s_wait_asynccnt 0x0" ::: "memory"); }

__device__ __forceinline__ uint32_t lds_addr32(const void* p) {
  return (uint32_t)(uintptr_t)p;
}

// A fragment (16x32, row-major source [rows, ld]):
// lane<16: M=lane, K 0..15 contiguous; lane>=16: M=lane-16, K 16..31 contiguous
__device__ __forceinline__ FragB load_a_frag(const uint16_t* mat,
                                             int m0, int ld, int k0, int lane) {
  FragB f;
  int row = m0 + (lane & 15);
  int c0  = k0 + ((lane & 16) ? 16 : 0);
  const uint16_t* p = mat + (size_t)row * ld + c0;
  f.q[0] = *(const uint4*)(p);
  f.q[1] = *(const uint4*)(p + 8);
  return f;
}

// B fragment (32x16). Source row-major [rows(=N), ld] holding B^T (row n = column n of B):
// lane<16: N=lane, K 0..15 contiguous; lane>=16: N=lane-16, K 16..31 contiguous
__device__ __forceinline__ FragB load_b_frag(const uint16_t* mat,
                                             int n0, int ld, int k0, int lane) {
  FragB f;
  int row = n0 + (lane & 15);
  int c0  = k0 + ((lane & 16) ? 16 : 0);
  const uint4* p = (const uint4*)(mat + (size_t)row * ld + c0);
  f.q[0] = p[0];
  f.q[1] = p[1];
  return f;
}

// R-window fragment with zero-fill for rows outside [0, KL)
__device__ __forceinline__ FragB load_r_frag(const uint16_t* __restrict__ Rn,
                                             int rbase, int k0, int lane) {
  int row = rbase + (lane & 15);
  bool ok = (row >= 0) && (row < KL);
  int rc = ok ? row : 0;
  int c0 = k0 + ((lane & 16) ? 16 : 0);
  const uint4* p = (const uint4*)(Rn + (size_t)rc * DH + c0);
  FragB f;
  f.q[0] = p[0];
  f.q[1] = p[1];
  if (!ok) { f.q[0] = make_uint4(0,0,0,0); f.q[1] = make_uint4(0,0,0,0); }
  return f;
}

// ---------------- conversion kernels ----------------
__global__ void cvt_bf16_kernel(const float* __restrict__ src,
                                uint16_t* __restrict__ dst, int n) {
  int i = blockIdx.x * 256 + threadIdx.x;
  if (i < n) dst[i] = f32_to_bf16_bits(src[i]);
}

// src [K,N] f32 -> dst [N,K] bf16 (transpose so GEMM B-fragments are contiguous)
__global__ void cvt_bf16_t_kernel(const float* __restrict__ src,
                                  uint16_t* __restrict__ dst, int K, int N) {
  int i = blockIdx.x * 256 + threadIdx.x;
  if (i < K * N) {
    int k = i / N, c = i - k * N;
    dst[(size_t)c * K + k] = f32_to_bf16_bits(src[i]);
  }
}

// ---------------- WMMA GEMM with async-LDS double buffering ----------------
// C[M,N] = A[M,K] * Bt[N,K]^T. Block = 128 threads (4 waves), 128x128 tile,
// each wave a 64x64 sub-tile. Per 32-deep k-step: stage A(128x32) and Bt(128x32)
// into LDS via global_load_async_to_lds_b128 (double buffered, ASYNCcnt).
__global__ __launch_bounds__(128) void gemm_bf16_nt(const uint16_t* __restrict__ A,
                                                    const uint16_t* __restrict__ Bt,
                                                    float* __restrict__ C,
                                                    int M, int N, int K) {
  __shared__ __align__(16) uint16_t Ast[2][128 * 32];
  __shared__ __align__(16) uint16_t Bst[2][128 * 32];

  int tid  = threadIdx.x;
  int lane = tid & 31;
  int wave = tid >> 5;
  int wm   = (wave >> 1) * 64;   // wave row offset in tile
  int wn   = (wave & 1) * 64;    // wave col offset in tile
  int m0   = blockIdx.y * 128;
  int n0   = blockIdx.x * 128;

  AccF acc[4][4];
#pragma unroll
  for (int i = 0; i < 4; i++)
#pragma unroll
    for (int j = 0; j < 4; j++)
#pragma unroll
      for (int r = 0; r < 8; r++) acc[i][j].f[r] = 0.0f;

  // stage: thread t copies row t (64B = 4x16B) of the A tile and of the Bt tile
  auto stage = [&](int k0, int buf) {
    const uint16_t* ga = A  + (size_t)(m0 + tid) * K + k0;
    const uint16_t* gb = Bt + (size_t)(n0 + tid) * K + k0;
    uint32_t la = lds_addr32(&Ast[buf][tid * 32]);
    uint32_t lb = lds_addr32(&Bst[buf][tid * 32]);
#pragma unroll
    for (int s = 0; s < 4; s++) {
      ASYNC_LD16(la + s * 16, (uint64_t)(uintptr_t)ga + s * 16);
      ASYNC_LD16(lb + s * 16, (uint64_t)(uintptr_t)gb + s * 16);
    }
  };

  stage(0, 0);
  for (int k0 = 0; k0 < K; k0 += 32) {
    int cur = (k0 >> 5) & 1;
    if (k0 + 32 < K) {
      stage(k0 + 32, cur ^ 1);
      wait_async_8();          // older (current-buffer) copies complete
    } else {
      wait_async_0();
    }
    __syncthreads();

    FragB a[4], b[4];
#pragma unroll
    for (int i = 0; i < 4; i++) a[i] = load_a_frag(&Ast[cur][0], wm + 16 * i, 32, 0, lane);
#pragma unroll
    for (int j = 0; j < 4; j++) b[j] = load_b_frag(&Bst[cur][0], wn + 16 * j, 32, 0, lane);
#pragma unroll
    for (int i = 0; i < 4; i++)
#pragma unroll
      for (int j = 0; j < 4; j++)
        acc[i][j].v = wmma_bf16(a[i], b[j], acc[i][j].v);

    __syncthreads();           // everyone done reading before overwrite
  }

  int half = (lane & 16) ? 8 : 0;
  int nl = lane & 15;
#pragma unroll
  for (int i = 0; i < 4; i++)
#pragma unroll
    for (int j = 0; j < 4; j++)
#pragma unroll
      for (int r = 0; r < 8; r++) {
        int row = m0 + wm + 16 * i + r + half;
        int col = n0 + wn + 16 * j + nl;
        C[(size_t)row * N + col] = acc[i][j].f[r];
      }
}

// ---------------- head split kernels ----------------
__global__ void split_q_kernel(const float* __restrict__ Hd,
                               const float* __restrict__ rwb,
                               const float* __restrict__ rrb,
                               uint16_t* __restrict__ Qw,
                               uint16_t* __restrict__ Qr) {
  int idx = blockIdx.x * 256 + threadIdx.x;
  if (idx >= BB * NH * QL * DH) return;
  int d = idx & 63;
  int i = (idx >> 6) & (QL - 1);
  int n = (idx >> 16) & (NH - 1);
  int b = idx >> 20;
  size_t src = ((size_t)((ML + i) * BB + b)) * (3 * DM) + n * DH + d;
  float q = Hd[src];
  size_t dst = (((size_t)(b * NH + n) * QL) + i) * DH + d;
  Qw[dst] = f32_to_bf16_bits(q + rwb[n * DH + d]);
  Qr[dst] = f32_to_bf16_bits(q + rrb[n * DH + d]);
}

__global__ void split_kv_kernel(const float* __restrict__ Hd,
                                uint16_t* __restrict__ Kb,   // [B,N,KL,DH]
                                uint16_t* __restrict__ Vt) { // [B,N,DH,KL] (transposed)
  int idx = blockIdx.x * 256 + threadIdx.x;
  if (idx >= BB * NH * KL * DH) return;
  int d = idx & 63;
  int j = (idx >> 6) & (KL - 1);
  int n = (idx >> 17) & (NH - 1);
  int b = idx >> 21;
  size_t row = (size_t)(j * BB + b) * (3 * DM);
  size_t kdst = (((size_t)(b * NH + n) * KL) + j) * DH + d;
  Kb[kdst] = f32_to_bf16_bits(Hd[row + DM + n * DH + d]);
  size_t vdst = ((size_t)(b * NH + n) * DH + d) * KL + j;
  Vt[vdst] = f32_to_bf16_bits(Hd[row + 2 * DM + n * DH + d]);
}

__global__ void split_r_kernel(const float* __restrict__ Rk, // [KL, DM]
                               uint16_t* __restrict__ Rr) {  // [N, KL, DH]
  int idx = blockIdx.x * 256 + threadIdx.x;
  if (idx >= NH * KL * DH) return;
  int d = idx & 63;
  int j = (idx >> 6) & (KL - 1);
  int n = idx >> 17;
  Rr[((size_t)n * KL + j) * DH + d] = f32_to_bf16_bits(Rk[(size_t)j * DM + n * DH + d]);
}

// ---------------- fused flash attention with rel-shift ----------------
// grid (QL/128, NH, BB), block 256 = 8 waves; each wave owns a 16-query block.
// K/V tiles staged once per workgroup via async copies (8x VMEM reduction),
// double-buffered on ASYNCcnt. Per-wave sliding R window for the rel-shift BD
// term (two WMMAs + ds_bpermute diagonal gather).
__global__ __launch_bounds__(256) void attn_kernel(const uint16_t* __restrict__ Qw,
                                                   const uint16_t* __restrict__ Qr,
                                                   const uint16_t* __restrict__ Kb,
                                                   const uint16_t* __restrict__ Vt,
                                                   const uint16_t* __restrict__ Rr,
                                                   uint16_t* __restrict__ AV) { // [QL*BB, DM]
  __shared__ __align__(16) uint16_t kstage[2][32 * 64];  // [j 32][dh 64]
  __shared__ __align__(16) uint16_t vstage[2][64 * 32];  // [dh 64][j 32]
  __shared__ __align__(16) uint16_t pstage[8][16 * 32];  // per-wave P staging

  int tid  = threadIdx.x;
  int wave = tid >> 5;
  int lane = tid & 31;
  int b = blockIdx.z, n = blockIdx.y;
  int ibase = blockIdx.x * 128;
  int i0 = ibase + wave * 16;

  const uint16_t* Qw_bn = Qw + ((size_t)(b * NH + n) * QL) * DH;
  const uint16_t* Qr_bn = Qr + ((size_t)(b * NH + n) * QL) * DH;
  const uint16_t* K_bn  = Kb + ((size_t)(b * NH + n) * KL) * DH;
  const uint16_t* V_bn  = Vt + ((size_t)(b * NH + n) * DH) * KL;
  const uint16_t* R_n   = Rr + ((size_t)n * KL) * DH;
  uint16_t* ps = &pstage[wave][0];

  // cooperative async stage of one 32-key step: K tile 4KB + V tile 4KB
  auto stage = [&](int j0, int buf) {
    {
      int row = tid >> 3, seg = tid & 7;             // 32 rows x 128B
      const uint16_t* g = K_bn + (size_t)(j0 + row) * DH + seg * 8;
      uint32_t l = lds_addr32(&kstage[buf][row * DH + seg * 8]);
      ASYNC_LD16(l, (uint64_t)(uintptr_t)g);
    }
    {
      int row = tid >> 2, seg = tid & 3;             // 64 rows x 64B
      const uint16_t* g = V_bn + (size_t)row * KL + j0 + seg * 8;
      uint32_t l = lds_addr32(&vstage[buf][row * 32 + seg * 8]);
      ASYNC_LD16(l, (uint64_t)(uintptr_t)g);
    }
  };

  // Q fragments (persistent): dh=64 -> two 32-deep A fragments per variant
  FragB qw0 = load_a_frag(Qw_bn, i0, DH, 0,  lane);
  FragB qw1 = load_a_frag(Qw_bn, i0, DH, 32, lane);
  FragB qr0 = load_a_frag(Qr_bn, i0, DH, 0,  lane);
  FragB qr1 = load_a_frag(Qr_bn, i0, DH, 32, lane);

  AccF O[4];
  float mrow[8], lrow[8];
#pragma unroll
  for (int dt = 0; dt < 4; dt++)
#pragma unroll
    for (int r = 0; r < 8; r++) O[dt].f[r] = 0.0f;
#pragma unroll
  for (int r = 0; r < 8; r++) { mrow[r] = NEG_BIG; lrow[r] = 0.0f; }

  // Sliding 32-row R window: BD[i,j] = Qr[i] . R[j - i + (QL-1)]
  int rbase = (QL - 1) - 15 - i0;   // window base for the j=0 tile
  FragB ra0 = load_r_frag(R_n, rbase, 0,  lane);
  FragB ra1 = load_r_frag(R_n, rbase, 32, lane);

  int nl   = lane & 15;
  int hb   = lane & 16;
  int half = hb ? 8 : 0;
  int jmax = ibase + 128 + ML;      // uniform across the workgroup (<= KL)

  stage(0, 0);
  for (int j0 = 0; j0 < jmax; j0 += 32) {
    int cur = (j0 >> 5) & 1;
    if (j0 + 32 < jmax) {
      stage(j0 + 32, cur ^ 1);
      wait_async_2();
    } else {
      wait_async_0();
    }
    __syncthreads();

    AccF S[2];
#pragma unroll
    for (int t = 0; t < 2; t++) {
      int j = j0 + 16 * t;
      // ---- AC: (q + r_w_bias) . K^T from LDS ----
      FragB kb0 = load_b_frag(&kstage[cur][0], 16 * t, DH, 0,  lane);
      FragB kb1 = load_b_frag(&kstage[cur][0], 16 * t, DH, 32, lane);
      f32x8 s;
      { AccF z;
#pragma unroll
        for (int r = 0; r < 8; r++) z.f[r] = 0.0f; s = z.v; }
      s = wmma_bf16(qw0, kb0, s);
      s = wmma_bf16(qw1, kb1, s);
      // ---- BD: banded window product, two WMMAs + lane gather for the shift ----
      FragB rb0 = load_r_frag(R_n, rbase + 16, 0,  lane);
      FragB rb1 = load_r_frag(R_n, rbase + 16, 32, lane);
      f32x8 ca, cb;
      { AccF z;
#pragma unroll
        for (int r = 0; r < 8; r++) z.f[r] = 0.0f; ca = z.v; cb = z.v; }
      ca = wmma_bf16(qr0, ra0, ca);
      ca = wmma_bf16(qr1, ra1, ca);
      cb = wmma_bf16(qr0, rb0, cb);
      cb = wmma_bf16(qr1, rb1, cb);
      AccF CA, CB, SV; CA.v = ca; CB.v = cb; SV.v = s;
#pragma unroll
      for (int r = 0; r < 8; r++) {
        int Mr = r + half;
        int c  = nl - Mr + 15;                 // 0..30
        int addr = (hb | (c & 15)) << 2;       // ds_bpermute byte address
        int ga = __builtin_amdgcn_ds_bpermute(addr, __float_as_int(CA.f[r]));
        int gb = __builtin_amdgcn_ds_bpermute(addr, __float_as_int(CB.f[r]));
        float bd = __int_as_float(c >= 16 ? gb : ga);
        float sv = (SV.f[r] + bd) * SCALE;
        bool masked = (j + nl) > (i0 + Mr + ML);
        S[t].f[r] = masked ? NEG_BIG : sv;
      }
      ra0 = rb0; ra1 = rb1; rbase += 16;       // slide window by 16
    }

    // ---- online softmax over this 16x32 score block ----
#pragma unroll
    for (int r = 0; r < 8; r++) {
      float tmax = fmaxf(S[0].f[r], S[1].f[r]);
#pragma unroll
      for (int off = 8; off >= 1; off >>= 1)
        tmax = fmaxf(tmax, __shfl_xor(tmax, off, 32));
      float mnew  = fmaxf(mrow[r], tmax);
      float scale = __expf(mrow[r] - mnew);
      float p0 = __expf(S[0].f[r] - mnew);
      float p1 = __expf(S[1].f[r] - mnew);
      float rs = p0 + p1;
#pragma unroll
      for (int off = 8; off >= 1; off >>= 1)
        rs += __shfl_xor(rs, off, 32);
      lrow[r] = lrow[r] * scale + rs;
      mrow[r] = mnew;
#pragma unroll
      for (int dt = 0; dt < 4; dt++) O[dt].f[r] *= scale;
      int row = r + half;
      ps[row * 32 + nl]      = f32_to_bf16_bits(p0);
      ps[row * 32 + 16 + nl] = f32_to_bf16_bits(p1);
    }

    // ---- PV: reload P as a 16x32 A fragment from LDS, V fragments from LDS ----
    FragB pA;
    {
      int row = lane & 15;
      int c0  = hb ? 16 : 0;
      const uint16_t* p = ps + row * 32 + c0;
      pA.q[0] = *(const uint4*)(p);
      pA.q[1] = *(const uint4*)(p + 8);
    }
#pragma unroll
    for (int dt = 0; dt < 4; dt++) {
      FragB vb = load_b_frag(&vstage[cur][0], dt * 16, 32, 0, lane);
      O[dt].v = wmma_bf16(pA, vb, O[dt].v);
    }
    __syncthreads();           // all waves done with `cur` before re-staging
  }

  // ---- normalize and store attn_vec rows in [q, b, n*dh] order (bf16) ----
#pragma unroll
  for (int r = 0; r < 8; r++) {
    float inv = 1.0f / lrow[r];
    int row = (i0 + r + half) * BB + b;
#pragma unroll
    for (int dt = 0; dt < 4; dt++) {
      int col = n * DH + dt * 16 + nl;
      AV[(size_t)row * DM + col] = f32_to_bf16_bits(O[dt].f[r] * inv);
    }
  }
}

// ---------------- residual + layernorm ----------------
__global__ __launch_bounds__(256) void resid_ln_kernel(const float* __restrict__ content,
                                                       const float* __restrict__ AO,
                                                       const float* __restrict__ gamma,
                                                       const float* __restrict__ beta,
                                                       float* __restrict__ out) {
  __shared__ float red[2][8];
  int row = blockIdx.x;   // q*BB rows, row = i*BB + b matches [q, b, d] flat
  int lane = threadIdx.x & 31;
  int wave = threadIdx.x >> 5;
  const float* c = content + (size_t)row * DM;
  const float* a = AO + (size_t)row * DM;
  float s = 0.0f, s2 = 0.0f;
  for (int d = threadIdx.x; d < DM; d += 256) {
    float x = c[d] + a[d];
    s += x; s2 += x * x;
  }
#pragma unroll
  for (int off = 16; off >= 1; off >>= 1) {
    s  += __shfl_xor(s,  off, 32);
    s2 += __shfl_xor(s2, off, 32);
  }
  if (lane == 0) { red[0][wave] = s; red[1][wave] = s2; }
  __syncthreads();
  float ts = 0.0f, ts2 = 0.0f;
#pragma unroll
  for (int w = 0; w < 8; w++) { ts += red[0][w]; ts2 += red[1][w]; }
  float mu  = ts * (1.0f / DM);
  float var = ts2 * (1.0f / DM) - mu * mu;
  float rstd = rsqrtf(var + 1e-5f);
  for (int d = threadIdx.x; d < DM; d += 256) {
    float x = c[d] + a[d];
    out[(size_t)row * DM + d] = (x - mu) * rstd * gamma[d] + beta[d];
  }
}

// ---------------- host launch ----------------
extern "C" void kernel_launch(void* const* d_in, const int* in_sizes, int n_in,
                              void* d_out, int out_size, void* d_ws, size_t ws_size,
                              hipStream_t stream) {
  const float* content = (const float*)d_in[0];   // [QL, BB, DM]
  const float* rel_pos = (const float*)d_in[1];   // [KL, DM]
  const float* mems    = (const float*)d_in[2];   // [ML, BB, DM]
  const float* rwb     = (const float*)d_in[3];   // [NH, DH]
  const float* rrb     = (const float*)d_in[4];   // [NH, DH]
  const float* W_qkv   = (const float*)d_in[5];   // [DM, 3*DM]
  const float* W_r     = (const float*)d_in[6];   // [DM, DM]
  const float* W_o     = (const float*)d_in[7];   // [DM, DM]
  const float* gamma   = (const float*)d_in[8];
  const float* beta    = (const float*)d_in[9];
  float* out = (float*)d_out;

  char* w = (char*)d_ws;
  size_t off = 0;
  auto alloc = [&](size_t bytes) -> void* {
    void* p = w + off;
    off = (off + bytes + 255) & ~(size_t)255;
    return p;
  };

  uint16_t* Xb     = (uint16_t*)alloc((size_t)KL * BB * DM * 2);   // cat bf16 [KL*BB, DM]
  uint16_t* Rpb    = (uint16_t*)alloc((size_t)KL * DM * 2);        // rel_pos bf16
  uint16_t* Wqkv_t = (uint16_t*)alloc((size_t)3 * DM * DM * 2);    // [3DM, DM]
  uint16_t* Wr_t   = (uint16_t*)alloc((size_t)DM * DM * 2);
  uint16_t* Wo_t   = (uint16_t*)alloc((size_t)DM * DM * 2);
  float*    Hd     = (float*)   alloc((size_t)KL * BB * 3 * DM * 4); // w_heads f32
  float*    Rk     = (float*)   alloc((size_t)KL * DM * 4);
  uint16_t* Qw     = (uint16_t*)alloc((size_t)BB * NH * QL * DH * 2);
  uint16_t* Qr     = (uint16_t*)alloc((size_t)BB * NH * QL * DH * 2);
  uint16_t* Kb     = (uint16_t*)alloc((size_t)BB * NH * KL * DH * 2);
  uint16_t* Vt     = (uint16_t*)alloc((size_t)BB * NH * DH * KL * 2);
  uint16_t* Rr     = (uint16_t*)alloc((size_t)NH * KL * DH * 2);
  uint16_t* AV     = (uint16_t*)alloc((size_t)QL * BB * DM * 2);
  float*    AO     = (float*)   alloc((size_t)QL * BB * DM * 4);
  (void)ws_size; (void)in_sizes; (void)n_in; (void)out_size;

  const int T = 256;
  // 1) bf16 conversions (mems || content -> cat) and transposed weights
  {
    int n1 = ML * BB * DM;
    cvt_bf16_kernel<<<(n1 + T - 1) / T, T, 0, stream>>>(mems, Xb, n1);
    int n2 = QL * BB * DM;
    cvt_bf16_kernel<<<(n2 + T - 1) / T, T, 0, stream>>>(content, Xb + (size_t)ML * BB * DM, n2);
    int n3 = KL * DM;
    cvt_bf16_kernel<<<(n3 + T - 1) / T, T, 0, stream>>>(rel_pos, Rpb, n3);
    int n4 = DM * 3 * DM;
    cvt_bf16_t_kernel<<<(n4 + T - 1) / T, T, 0, stream>>>(W_qkv, Wqkv_t, DM, 3 * DM);
    int n5 = DM * DM;
    cvt_bf16_t_kernel<<<(n5 + T - 1) / T, T, 0, stream>>>(W_r, Wr_t, DM, DM);
    cvt_bf16_t_kernel<<<(n5 + T - 1) / T, T, 0, stream>>>(W_o, Wo_t, DM, DM);
  }
  // 2) w_heads = cat @ W_qkv  (8192 x 3072 x 1024)
  gemm_bf16_nt<<<dim3(3 * DM / 128, KL * BB / 128), 128, 0, stream>>>(Xb, Wqkv_t, Hd,
                                                                      KL * BB, 3 * DM, DM);
  // 3) r_k = rel_pos @ W_r  (2048 x 1024 x 1024)
  gemm_bf16_nt<<<dim3(DM / 128, KL / 128), 128, 0, stream>>>(Rpb, Wr_t, Rk, KL, DM, DM);
  // 4) head splits (+biases, V transpose)
  {
    int nq = BB * NH * QL * DH;
    split_q_kernel<<<(nq + T - 1) / T, T, 0, stream>>>(Hd, rwb, rrb, Qw, Qr);
    int nkv = BB * NH * KL * DH;
    split_kv_kernel<<<(nkv + T - 1) / T, T, 0, stream>>>(Hd, Kb, Vt);
    int nr = NH * KL * DH;
    split_r_kernel<<<(nr + T - 1) / T, T, 0, stream>>>(Rk, Rr);
  }
  // 5) fused flash attention with Transformer-XL rel-shift
  attn_kernel<<<dim3(QL / 128, NH, BB), 256, 0, stream>>>(Qw, Qr, Kb, Vt, Rr, AV);
  // 6) attn_out = attn_vec @ W_o  (4096 x 1024 x 1024)
  gemm_bf16_nt<<<dim3(DM / 128, QL * BB / 128), 128, 0, stream>>>(AV, Wo_t, AO,
                                                                  QL * BB, DM, DM);
  // 7) residual + layernorm
  resid_ln_kernel<<<QL * BB, 256, 0, stream>>>(content, AO, gamma, beta, out);
}
